// Decoder_v2_21603685499559
// MI455X (gfx1250) — matrix operand order repaired and checked by reading, not compiled, not run on hardware
//
#include <hip/hip_runtime.h>
#include <hip/hip_bf16.h>

typedef __attribute__((ext_vector_type(16))) _Float16 v16h;
typedef __attribute__((ext_vector_type(8)))  _Float16 v8h;
typedef __attribute__((ext_vector_type(8)))  float    v8f;

union AFrag { v16h v; v8h h[2]; };

__device__ __forceinline__ v8f wmma_f16(v16h a, v16h b, v8f c) {
  return __builtin_amdgcn_wmma_f32_16x16x32_f16(false, a, false, b, (short)0, c, false, false);
}

// ---------------------------------------------------------------------------
// Prep 1: convert layer weights to f16, transposed to [o][u] (N-major) with
// the reference's normalization folded in. Makes every WMMA B-fragment a
// single contiguous 32B load.
// ---------------------------------------------------------------------------
__global__ void prep_weights(const float* __restrict__ la_W0,
                             const float* __restrict__ la_W1,
                             const float* __restrict__ la_W2,
                             _Float16* __restrict__ w0t,
                             _Float16* __restrict__ w1t,
                             _Float16* __restrict__ w2t) {
  const int N0 = 10 * 224 * 352;
  const int N1 = 10 * 64 * 64;
  const int N2 = 10 * 32 * 32;
  const int total = N0 + N1 + N2;
  for (int idx = blockIdx.x * blockDim.x + threadIdx.x; idx < total;
       idx += gridDim.x * blockDim.x) {
    if (idx < N0) {
      int l = idx / (224 * 352);
      int rem = idx - l * 224 * 352;
      int o = rem / 352;
      int u = rem - o * 352;
      w0t[idx] = (_Float16)(la_W0[(l * 352 + u) * 224 + o] * 0.05330017909f); // 1/sqrt(352)
    } else if (idx < N0 + N1) {
      int i = idx - N0;
      int l = i / 4096;
      int rem = i - l * 4096;
      int o = rem / 64;
      int u = rem - o * 64;
      w1t[i] = (_Float16)(la_W1[(l * 64 + u) * 64 + o] * 0.125f); // 1/sqrt(64)
    } else {
      int i = idx - N0 - N1;
      int l = i / 1024;
      int rem = i - l * 1024;
      int o = rem / 32;
      int u = rem - o * 32;
      w2t[i] = (_Float16)(la_W2[(l * 32 + u) * 32 + o] * 0.17677669529f); // 1/sqrt(32)
    }
  }
}

// ---------------------------------------------------------------------------
// Prep 2: collapse the input stage. z_so3 only varies per batch (4), and the
// query enters only through 9 spherical harmonic coefficients, so the whole
// p0..p5 / W0s / W0v / W0t stage folds into per-batch tensors:
//   s[o]      = b0[o] + sum_j Sx[b][o][j]            * sph[1+j]
//   v[o][i]   =         sum_m Vx[b][o][m][i]         * sph[m]      (m<9)
//   t[o][m5]  =         sum_m8 Tx[b][o][m8][m5]      * sph[1+m8]   (m8<8)
// ---------------------------------------------------------------------------
__global__ void prep_batch(const float* __restrict__ z_so3,
                           const float* __restrict__ w_dtp,
                           const float* __restrict__ W0s,
                           const float* __restrict__ W0v,
                           const float* __restrict__ W0t,
                           float* __restrict__ Sx,
                           float* __restrict__ Vx,
                           float* __restrict__ Tx) {
  __shared__ float C111[3][3][3];
  __shared__ float C112[3][3][5];
  __shared__ float C121[3][5][3];
  __shared__ float C122[3][5][5];
  if (threadIdx.x == 0) {
    float EPS[3][3][3];
    float B[5][3][3];
    for (int i = 0; i < 3; ++i)
      for (int j = 0; j < 3; ++j)
        for (int k = 0; k < 3; ++k) EPS[i][j][k] = 0.0f;
    for (int k = 0; k < 5; ++k)
      for (int i = 0; i < 3; ++i)
        for (int j = 0; j < 3; ++j) B[k][i][j] = 0.0f;
    EPS[0][1][2] = EPS[1][2][0] = EPS[2][0][1] = 1.0f;
    EPS[0][2][1] = EPS[2][1][0] = EPS[1][0][2] = -1.0f;
    const float r2 = 0.70710678119f;       // 1/sqrt(2)
    const float i6 = 0.40824829046f;       // 1/sqrt(6)
    B[0][0][1] = B[0][1][0] = r2;
    B[1][1][2] = B[1][2][1] = r2;
    B[3][0][2] = B[3][2][0] = r2;
    B[2][0][0] = -i6; B[2][1][1] = -i6; B[2][2][2] = 2.0f * i6;
    B[4][0][0] = r2;  B[4][1][1] = -r2;
    for (int i = 0; i < 3; ++i)
      for (int j = 0; j < 3; ++j)
        for (int k = 0; k < 3; ++k) C111[i][j][k] = EPS[i][j][k] * r2;
    for (int i = 0; i < 3; ++i)
      for (int j = 0; j < 3; ++j)
        for (int k = 0; k < 5; ++k) C112[i][j][k] = B[k][i][j];
    for (int i = 0; i < 3; ++i)
      for (int k = 0; k < 5; ++k)
        for (int j = 0; j < 3; ++j) C121[i][k][j] = B[k][i][j] * 0.77459666924f; // sqrt(0.6)
    float M[3][5][5];
    float nrm = 0.0f;
    for (int i = 0; i < 3; ++i)
      for (int k = 0; k < 5; ++k)
        for (int lq = 0; lq < 5; ++lq) {
          float acc = 0.0f;
          for (int a = 0; a < 3; ++a)
            for (int bb = 0; bb < 3; ++bb)
              for (int c = 0; c < 3; ++c)
                acc += EPS[i][a][bb] * B[k][a][c] * B[lq][c][bb];
          M[i][k][lq] = acc;
          nrm += acc * acc;
        }
    const float sc = 2.2360679775f / sqrtf(nrm); // sqrt(5)/||M||
    for (int i = 0; i < 3; ++i)
      for (int k = 0; k < 5; ++k)
        for (int lq = 0; lq < 5; ++lq) C122[i][k][lq] = M[i][k][lq] * sc;
  }
  __syncthreads();

  // Sx[b][o][j] (4,128,3)
  for (int idx = threadIdx.x; idx < 4 * 128 * 3; idx += blockDim.x) {
    int b = idx / 384;
    int rem = idx - b * 384;
    int o = rem / 3;
    int j = rem - o * 3;
    float acc = 0.0f;
    for (int u = 0; u < 256; ++u)
      acc += w_dtp[256 + u] * z_so3[(b * 256 + u) * 3 + j] * W0s[u * 128 + o];
    Sx[idx] = acc * (0.57735026919f / 16.0f); // 1/(sqrt3 * sqrt256)
  }
  // Vx[b][o][m][i] (4,64,9,3)
  for (int idx = threadIdx.x; idx < 4 * 64 * 27; idx += blockDim.x) {
    int b = idx / (64 * 27);
    int rem = idx - b * 64 * 27;
    int o = rem / 27;
    int mi = rem - o * 27;
    int m = mi / 3;
    int i = mi - m * 3;
    float acc = 0.0f;
    if (m == 0) {
      for (int u = 0; u < 256; ++u)
        acc += w_dtp[u] * z_so3[(b * 256 + u) * 3 + i] * W0v[u * 64 + o];
    } else if (m < 4) {
      int j = m - 1;
      for (int u = 0; u < 256; ++u) {
        float d = 0.0f;
        for (int a = 0; a < 3; ++a) d += z_so3[(b * 256 + u) * 3 + a] * C111[a][j][i];
        acc += w_dtp[2 * 256 + u] * d * W0v[(256 + u) * 64 + o];
      }
    } else {
      int k = m - 4;
      for (int u = 0; u < 256; ++u) {
        float d = 0.0f;
        for (int a = 0; a < 3; ++a) d += z_so3[(b * 256 + u) * 3 + a] * C121[a][k][i];
        acc += w_dtp[4 * 256 + u] * d * W0v[(512 + u) * 64 + o];
      }
    }
    Vx[idx] = acc * 0.036084392f; // 1/sqrt(768)
  }
  // Tx[b][o][m8][m5] (4,32,8,5)
  for (int idx = threadIdx.x; idx < 4 * 32 * 40; idx += blockDim.x) {
    int b = idx / (32 * 40);
    int rem = idx - b * 32 * 40;
    int o = rem / 40;
    int mm = rem - o * 40;
    int m8 = mm / 5;
    int m5 = mm - m8 * 5;
    float acc = 0.0f;
    if (m8 < 3) {
      int j = m8;
      for (int u = 0; u < 256; ++u) {
        float d = 0.0f;
        for (int a = 0; a < 3; ++a) d += z_so3[(b * 256 + u) * 3 + a] * C112[a][j][m5];
        acc += w_dtp[3 * 256 + u] * d * W0t[u * 32 + o];
      }
    } else {
      int k = m8 - 3;
      for (int u = 0; u < 256; ++u) {
        float d = 0.0f;
        for (int a = 0; a < 3; ++a) d += z_so3[(b * 256 + u) * 3 + a] * C122[a][k][m5];
        acc += w_dtp[5 * 256 + u] * d * W0t[(256 + u) * 32 + o];
      }
    }
    Tx[idx] = acc * 0.044194174f; // 1/sqrt(512)
  }
}

// ---------------------------------------------------------------------------
// Main kernel: one wave32 per 16-point tile. State f32 in LDS, f16 staging,
// all three per-layer matmuls via v_wmma_f32_16x16x32_f16.
// ---------------------------------------------------------------------------
__global__ __launch_bounds__(32) void decoder_main(
    const float* __restrict__ query,
    const float* __restrict__ b0,
    const float* __restrict__ la_w,
    const float* __restrict__ la_b,
    const float* __restrict__ inv_w,
    const float* __restrict__ fc_W,
    const float* __restrict__ fc_b,
    const _Float16* __restrict__ w0t,
    const _Float16* __restrict__ w1t,
    const _Float16* __restrict__ w2t,
    const float* __restrict__ Sx,
    const float* __restrict__ Vx,
    const float* __restrict__ Tx,
    float* __restrict__ out) {
  extern __shared__ char smem[];
  float* sph    = (float*)smem;              // 16*12 f32        768 B
  float* gBuf   = sph + 16 * 12;             // 16*96 f32       6144 B
  float* sState = gBuf + 16 * 96;            // 16*128 f32      8192 B
  float* vState = sState + 16 * 128;         // 3*16*64 f32    12288 B
  float* tState = vState + 3 * 16 * 64;      // 5*16*32 f32    10240 B
  float* sTmp   = tState + 5 * 16 * 32;      // 16*128 f32      8192 B
  float* vTmp   = sTmp + 16 * 128;           // 3*16*64 f32    12288 B
  float* tTmp   = vTmp + 3 * 16 * 64;        // 5*16*32 f32    10240 B
  _Float16* siH = (_Float16*)(tTmp + 5 * 16 * 32); // 16*352 f16 11264 B
  _Float16* vHs = siH + 16 * 352;            // 3*16*64 f16     6144 B
  _Float16* tHs = vHs + 3 * 16 * 64;         // 5*16*32 f16     5120 B
                                             // total          90880 B

  const int tid  = threadIdx.x;
  const int lm   = tid & 15;
  const int half = tid >> 4;
  const int n0   = blockIdx.x * 16;
  const int batch = n0 >> 12; // 4096 points per batch

  // ---- spherical harmonics (9 coeffs), lanes 0..15 = points ----
  if (tid < 16) {
    const int n = n0 + tid;
    float qx = query[n * 3 + 0], qy = query[n * 3 + 1], qz = query[n * 3 + 2];
    float r = sqrtf(qx * qx + qy * qy + qz * qz);
    float inr = 1.0f / r;
    float ux = qx * inr, uy = qy * inr, uz = qz * inr;
    float* sp = sph + tid * 12;
    sp[0] = r;
    sp[1] = 1.7320508076f * qx;
    sp[2] = 1.7320508076f * qy;
    sp[3] = 1.7320508076f * qz;
    sp[4] = 3.8729833462f * ux * uy * r;
    sp[5] = 3.8729833462f * uy * uz * r;
    sp[6] = 1.1180339887f * (3.0f * uz * uz - 1.0f) * r;
    sp[7] = 3.8729833462f * ux * uz * r;
    sp[8] = 1.9364916731f * (ux * ux - uy * uy) * r;
  }
  __syncthreads();

  // ---- initial state from collapsed per-batch tensors ----
  {
    const float* SxB = Sx + batch * 384;
    for (int idx = tid; idx < 2048; idx += 32) {
      int p = idx >> 7, o = idx & 127;
      const float* sp = sph + p * 12;
      float acc = b0[o];
      for (int j = 0; j < 3; ++j) acc += SxB[o * 3 + j] * sp[1 + j];
      sState[idx] = acc;
    }
    const float* VxB = Vx + batch * 1728;
    for (int idx = tid; idx < 3072; idx += 32) {
      int comp = idx >> 10, rem = idx & 1023, p = rem >> 6, o = rem & 63;
      const float* sp = sph + p * 12;
      float acc = 0.0f;
      for (int m = 0; m < 9; ++m) acc += VxB[o * 27 + m * 3 + comp] * sp[m];
      vState[idx] = acc;
    }
    const float* TxB = Tx + batch * 1280;
    for (int idx = tid; idx < 2560; idx += 32) {
      int comp = idx >> 9, rem = idx & 511, p = rem >> 5, o = rem & 31;
      const float* sp = sph + p * 12;
      float acc = 0.0f;
      for (int m = 0; m < 8; ++m) acc += TxB[o * 40 + m * 5 + comp] * sp[1 + m];
      tState[idx] = acc;
    }
  }
  __syncthreads();

  // ---- 10 linear-activation layers (5 residual pairs) ----
  for (int l = 0; l < 10; ++l) {
    const bool resid = (l & 1) != 0;  // odd sub-layer: state += f(tmp)
    const float* srcS = resid ? sTmp : sState;
    const float* srcV = resid ? vTmp : vState;
    const float* srcT = resid ? tTmp : tState;
    float* dstS = resid ? sState : sTmp;
    float* dstV = resid ? vState : vTmp;
    float* dstT = resid ? tState : tTmp;
    const float* lw = la_w + l * 224;
    const float* lb = la_b + l * 224;

    if (l < 9) __builtin_prefetch(w0t + (l + 1) * 224 * 352, 0, 0);

    // stage si = [la_w * invariants | s] as f16
    for (int idx = tid; idx < 16 * 352; idx += 32) {
      int p = idx / 352, k = idx - p * 352;
      float val;
      if (k < 128) {
        float s = srcS[p * 128 + k];
        val = lw[k] * s * s;
      } else if (k < 192) {
        int c = k - 128;
        float a0 = srcV[p * 64 + c];
        float a1 = srcV[1024 + p * 64 + c];
        float a2 = srcV[2048 + p * 64 + c];
        val = lw[k] * (a0 * a0 + a1 * a1 + a2 * a2) * 0.57735026919f;
      } else if (k < 224) {
        int c = k - 192;
        float acc = 0.0f;
        for (int m5 = 0; m5 < 5; ++m5) {
          float a = srcT[m5 * 512 + p * 32 + c];
          acc += a * a;
        }
        val = lw[k] * acc * 0.44721359550f;
      } else {
        val = srcS[p * 128 + (k - 224)];
      }
      siH[idx] = (_Float16)val;
    }
    for (int idx = tid; idx < 3072; idx += 32) vHs[idx] = (_Float16)srcV[idx];
    for (int idx = tid; idx < 2560; idx += 32) tHs[idx] = (_Float16)srcT[idx];
    __syncthreads();

    // ---- so = si @ W0 : M=16, K=352 (11 steps), N=224 (14 tiles) ----
    const _Float16* W0l = w0t + l * 224 * 352;
    for (int nt = 0; nt < 14; ++nt) {
      v8f acc = {};
      const int col = nt * 16 + lm;
      const _Float16* wb = W0l + col * 352 + half * 16;
      const _Float16* ar = siH + lm * 352 + half * 16;
      for (int kb = 0; kb < 11; ++kb) {
        AFrag a;
        a.h[0] = *(const v8h*)(ar + kb * 32);
        a.h[1] = *(const v8h*)(ar + kb * 32 + 8);
        v16h b = *(const v16h*)(wb + kb * 32);
        acc = wmma_f16(a.v, b, acc);
      }
      const int o = nt * 16 + lm;
      const float bias = lb[o];
      if (o < 128) { // uniform per nt (nt<8)
        for (int r = 0; r < 8; ++r) {
          int m = half * 8 + r;
          float x = acc[r] + bias;
          float act = x / (1.0f + __expf(-x)); // silu
          if (resid) dstS[m * 128 + o] += act;
          else       dstS[m * 128 + o] = act;
        }
      } else {
        for (int r = 0; r < 8; ++r) {
          int m = half * 8 + r;
          float x = acc[r] + bias;
          gBuf[m * 96 + (o - 128)] = 1.0f / (1.0f + __expf(-x)); // sigmoid gate
        }
      }
    }
    __syncthreads();

    // ---- vo = v @ W1 (per xyz comp): M=16, K=64 (2 steps), N=64 (4 tiles) ----
    const _Float16* W1l = w1t + l * 4096;
    for (int comp = 0; comp < 3; ++comp) {
      for (int nt = 0; nt < 4; ++nt) {
        v8f acc = {};
        const int col = nt * 16 + lm;
        const _Float16* wb = W1l + col * 64 + half * 16;
        const _Float16* ar = vHs + comp * 1024 + lm * 64 + half * 16;
        for (int kb = 0; kb < 2; ++kb) {
          AFrag a;
          a.h[0] = *(const v8h*)(ar + kb * 32);
          a.h[1] = *(const v8h*)(ar + kb * 32 + 8);
          v16h b = *(const v16h*)(wb + kb * 32);
          acc = wmma_f16(a.v, b, acc);
        }
        const int o = nt * 16 + lm;
        for (int r = 0; r < 8; ++r) {
          int m = half * 8 + r;
          float val = acc[r] * gBuf[m * 96 + o];
          int di = comp * 1024 + m * 64 + o;
          if (resid) dstV[di] += val;
          else       dstV[di] = val;
        }
      }
    }

    // ---- to = t @ W2 (per 5 comps): M=16, K=32 (1 step), N=32 (2 tiles) ----
    const _Float16* W2l = w2t + l * 1024;
    for (int comp = 0; comp < 5; ++comp) {
      for (int nt = 0; nt < 2; ++nt) {
        v8f acc = {};
        const int col = nt * 16 + lm;
        const _Float16* ar = tHs + comp * 512 + lm * 32 + half * 16;
        AFrag a;
        a.h[0] = *(const v8h*)(ar);
        a.h[1] = *(const v8h*)(ar + 8);
        v16h b = *(const v16h*)(W2l + col * 32 + half * 16);
        acc = wmma_f16(a.v, b, acc);
        const int o = nt * 16 + lm;
        for (int r = 0; r < 8; ++r) {
          int m = half * 8 + r;
          float val = acc[r] * gBuf[m * 96 + 64 + o];
          int di = comp * 512 + m * 32 + o;
          if (resid) dstT[di] += val;
          else       dstT[di] = val;
        }
      }
    }
    __syncthreads();
  }

  // ---- final invariant head ----
  if (tid < 16) {
    const int p = tid;
    float acc = fc_b[0];
    for (int k = 0; k < 224; ++k) {
      float x;
      if (k < 128) {
        float s = sState[p * 128 + k];
        x = s * s;
      } else if (k < 192) {
        int c = k - 128;
        float a0 = vState[p * 64 + c];
        float a1 = vState[1024 + p * 64 + c];
        float a2 = vState[2048 + p * 64 + c];
        x = (a0 * a0 + a1 * a1 + a2 * a2) * 0.57735026919f;
      } else {
        int c = k - 192;
        float s2 = 0.0f;
        for (int m5 = 0; m5 < 5; ++m5) {
          float a = tState[m5 * 512 + p * 32 + c];
          s2 += a * a;
        }
        x = s2 * 0.44721359550f;
      }
      x *= inv_w[k];
      if (x > 0.0f) acc += fc_W[k] * x;
    }
    out[n0 + p] = acc;
  }
}

// ---------------------------------------------------------------------------
extern "C" void kernel_launch(void* const* d_in, const int* in_sizes, int n_in,
                              void* d_out, int out_size, void* d_ws, size_t ws_size,
                              hipStream_t stream) {
  (void)in_sizes; (void)n_in; (void)out_size; (void)ws_size;
  const float* z_so3 = (const float*)d_in[0];
  const float* query = (const float*)d_in[1];
  const float* w_dtp = (const float*)d_in[2];
  const float* W0s   = (const float*)d_in[3];
  const float* W0v   = (const float*)d_in[4];
  const float* W0t   = (const float*)d_in[5];
  const float* b0    = (const float*)d_in[6];
  const float* la_w  = (const float*)d_in[7];
  const float* la_W0 = (const float*)d_in[8];
  const float* la_W1 = (const float*)d_in[9];
  const float* la_W2 = (const float*)d_in[10];
  const float* la_b  = (const float*)d_in[11];
  const float* inv_w = (const float*)d_in[12];
  const float* fc_W  = (const float*)d_in[13];
  const float* fc_b  = (const float*)d_in[14];
  float* out = (float*)d_out;

  // workspace layout (~1.74 MB)
  _Float16* w0t = (_Float16*)d_ws;        // 10*224*352 halves
  _Float16* w1t = w0t + 788480;           // 10*64*64
  _Float16* w2t = w1t + 40960;            // 10*32*32
  float* Sx = (float*)(w2t + 10240);      // 4*128*3
  float* Vx = Sx + 1536;                  // 4*64*27
  float* Tx = Vx + 6912;                  // 4*32*40

  prep_weights<<<512, 256, 0, stream>>>(la_W0, la_W1, la_W2, w0t, w1t, w2t);
  prep_batch<<<1, 256, 0, stream>>>(z_so3, w_dtp, W0s, W0v, W0t, Sx, Vx, Tx);

  const int smem_bytes = 90880;
  (void)hipFuncSetAttribute((const void*)decoder_main,
                            hipFuncAttributeMaxDynamicSharedMemorySize, smem_bytes);
  decoder_main<<<1024, 32, smem_bytes, stream>>>(query, b0, la_w, la_b, inv_w, fc_W, fc_b,
                                                 w0t, w1t, w2t, Sx, Vx, Tx, out);
}
